// MambaBlock_62448824483850
// MI455X (gfx1250) — compile-verified, matrix operand order checked by reading, not run on hardware
//
#include <hip/hip_runtime.h>
#include <hip/hip_bf16.h>

typedef __bf16 bf16_t;
typedef __attribute__((ext_vector_type(16))) __bf16 v16bf;
typedef __attribute__((ext_vector_type(8)))  float  v8f;

union FragBF { unsigned int u[8]; v16bf v; };

#define D_MODEL 1024
#define D_STATE 16
#define D_CONV  4
#define D_INNER 2048
#define BATCH   2
#define SEQLEN  1024
#define TOKENS  (BATCH * SEQLEN)   // 2048

// async copy: one wave instruction moves 32 lanes x 16B global -> LDS
__device__ __forceinline__ void async_ld_b128(unsigned lds_byte_addr,
                                              unsigned long long gaddr) {
    asm volatile("global_load_async_to_lds_b128 %0, %1, off"
                 :: "v"(lds_byte_addr), "v"(gaddr) : "memory");
}

// ---------------------------------------------------------------------------
// f32 -> bf16 cast
// ---------------------------------------------------------------------------
__global__ void cvt_f32_bf16(const float* __restrict__ in, bf16_t* __restrict__ out, int n) {
    int i = blockIdx.x * blockDim.x + threadIdx.x;
    if (i < n) out[i] = (bf16_t)in[i];
}

// ---------------------------------------------------------------------------
// C[m][n] = sum_k A[m*K+k] * B[n*K+k]   (A: MxK bf16, B: NxK bf16, C: MxN f32)
// 128 threads = 4 waves; block tile 64x64; wave tile 32x32 (2x2 WMMA).
// K stepped by 32 through double-buffered LDS filled with async-to-LDS copies.
// Last iteration peeled so the steady-state loop is branch-free.
// ---------------------------------------------------------------------------
__global__ __launch_bounds__(128) void gemm_bf16_wmma(
    const bf16_t* __restrict__ A, const bf16_t* __restrict__ Bm,
    float* __restrict__ C, int M, int N, int K)
{
    __shared__ __align__(16) bf16_t lA[2][64 * 32];
    __shared__ __align__(16) bf16_t lB[2][64 * 32];

    const int tid      = threadIdx.x;
    const int lane     = tid & 31;
    const int wave     = tid >> 5;
    const int waveM    = (wave >> 1) * 32;
    const int waveN    = (wave & 1) * 32;
    const int blockM   = blockIdx.y * 64;
    const int blockN   = blockIdx.x * 64;
    const int laneHalf = lane >> 4;      // 0: lanes 0-15, 1: lanes 16-31
    const int lm       = lane & 15;
    const int kb0      = laneHalf * 8;   // per documented 16-bit fragment layout

    // this thread's two 16B chunks within each 64x32 tile
    const int e0 = tid * 8;
    const int e1 = (tid + 128) * 8;
    const int r0 = e0 >> 5, c0 = e0 & 31;
    const int r1 = e1 >> 5, c1 = e1 & 31;
    const bf16_t* gA0 = A  + (size_t)(blockM + r0) * K + c0;
    const bf16_t* gA1 = A  + (size_t)(blockM + r1) * K + c1;
    const bf16_t* gB0 = Bm + (size_t)(blockN + r0) * K + c0;
    const bf16_t* gB1 = Bm + (size_t)(blockN + r1) * K + c1;

    auto issue_tile = [&](int buf, int k0) {
        async_ld_b128((unsigned)(uintptr_t)&lA[buf][e0], (unsigned long long)(uintptr_t)(gA0 + k0));
        async_ld_b128((unsigned)(uintptr_t)&lA[buf][e1], (unsigned long long)(uintptr_t)(gA1 + k0));
        async_ld_b128((unsigned)(uintptr_t)&lB[buf][e0], (unsigned long long)(uintptr_t)(gB0 + k0));
        async_ld_b128((unsigned)(uintptr_t)&lB[buf][e1], (unsigned long long)(uintptr_t)(gB1 + k0));
    };

    v8f acc[2][2] = {};

    auto compute_tile = [&](int buf) {
        const unsigned int* uA = (const unsigned int*)lA[buf];
        const unsigned int* uB = (const unsigned int*)lB[buf];
        FragBF a[2], b[2];
        #pragma unroll
        for (int s = 0; s < 2; ++s) {
            int rowA = waveM + s * 16 + lm;
            int rowB = waveN + s * 16 + lm;
            #pragma unroll
            for (int j = 0; j < 4; ++j) {
                int k  = kb0 + 2 * j;        // K 0..15 half
                int k2 = 16 + kb0 + 2 * j;   // K 16..31 half
                a[s].u[j]     = uA[rowA * 16 + (k  >> 1)];
                a[s].u[j + 4] = uA[rowA * 16 + (k2 >> 1)];
                b[s].u[j]     = uB[rowB * 16 + (k  >> 1)];
                b[s].u[j + 4] = uB[rowB * 16 + (k2 >> 1)];
            }
        }
        #pragma unroll
        for (int i = 0; i < 2; ++i)
            #pragma unroll
            for (int j = 0; j < 2; ++j)
                acc[i][j] = __builtin_amdgcn_wmma_f32_16x16x32_bf16(
                    false, a[i].v, false, b[j].v, (short)0, acc[i][j], false, false);
    };

    const int nTiles = K >> 5;
    issue_tile(0, 0);
    // steady state: branch-free issue(next) -> wait -> barrier -> compute -> barrier
    for (int t = 0; t < nTiles - 1; ++t) {
        const int cur = t & 1;
        issue_tile(cur ^ 1, (t + 1) * 32);               // prefetch next tile
        asm volatile("s_wait_asynccnt 0x4" ::: "memory"); // tile t has landed
        __syncthreads();
        compute_tile(cur);
        __syncthreads();   // all reads of buf[cur] done before it is refilled
    }
    // peeled final tile
    asm volatile("s_wait_asynccnt 0x0" ::: "memory");
    __syncthreads();
    compute_tile((nTiles - 1) & 1);

    // C/D layout: lane n = lane&15; VGPR r -> M = r + 8*(lane>=16)
    const int mAdd = laneHalf * 8;
    #pragma unroll
    for (int i = 0; i < 2; ++i)
        #pragma unroll
        for (int j = 0; j < 2; ++j)
            #pragma unroll
            for (int r = 0; r < 8; ++r) {
                int m  = blockM + waveM + i * 16 + mAdd + r;
                int nn = blockN + waveN + j * 16 + lm;
                C[(size_t)m * N + nn] = acc[i][j][r];
            }
}

// ---------------------------------------------------------------------------
// xc = silu(conv_b + depthwise-causal-conv4(x_ssm));  x_ssm = xz[..., :2048]
// ---------------------------------------------------------------------------
__global__ void conv_silu_kernel(const float* __restrict__ xz,
                                 const float* __restrict__ conv_w,
                                 const float* __restrict__ conv_b,
                                 float* __restrict__ xc)
{
    int idx = blockIdx.x * blockDim.x + threadIdx.x;
    if (idx >= BATCH * SEQLEN * D_INNER) return;
    int d   = idx % D_INNER;
    int t   = idx / D_INNER;        // token = b*SEQLEN + l
    int l   = t % SEQLEN;
    float v = conv_b[d];
    #pragma unroll
    for (int k = 0; k < D_CONV; ++k) {
        int ls = l + k - (D_CONV - 1);
        if (ls >= 0)
            v += conv_w[d * D_CONV + k] * xz[(size_t)(t + k - (D_CONV - 1)) * (2 * D_INNER) + d];
    }
    v = v / (1.0f + __expf(-v));    // silu
    xc[(size_t)t * D_INNER + d] = v;
}

// ---------------------------------------------------------------------------
// Per token: xp = x_proj_w (33 x 2048) @ xc_row  -> dt_r scalar, B[16], C[16]
// one block (256 thr = 8 waves) per token, xc row staged in LDS
// ---------------------------------------------------------------------------
__global__ __launch_bounds__(256) void xproj_kernel(const float* __restrict__ xc,
                                                    const float* __restrict__ x_proj_w,
                                                    float* __restrict__ dtr,
                                                    float* __restrict__ Bv,
                                                    float* __restrict__ Cv)
{
    __shared__ float sxc[D_INNER];
    const int t    = blockIdx.x;
    const int tid  = threadIdx.x;
    const int lane = tid & 31;
    const int wave = tid >> 5;

    for (int i = tid; i < D_INNER; i += 256)
        sxc[i] = xc[(size_t)t * D_INNER + i];
    __syncthreads();

    for (int j = wave; j < 2 * D_STATE + 1; j += 8) {
        float p = 0.0f;
        const float* wrow = &x_proj_w[(size_t)j * D_INNER];
        for (int i = lane; i < D_INNER; i += 32)
            p += wrow[i] * sxc[i];
        #pragma unroll
        for (int m = 16; m >= 1; m >>= 1)
            p += __shfl_xor(p, m, 32);
        if (lane == 0) {
            if (j == 0)                dtr[t] = p;
            else if (j <= D_STATE)     Bv[(size_t)t * D_STATE + (j - 1)] = p;
            else                       Cv[(size_t)t * D_STATE + (j - 1 - D_STATE)] = p;
        }
    }
}

// ---------------------------------------------------------------------------
// Selective scan. One 16-lane group per channel (b,d); lane n holds state n.
// Fused epilogue: y = (sum_n h*C) + xc*D, y *= silu(z), store bf16 for GEMM2.
// ---------------------------------------------------------------------------
__global__ __launch_bounds__(128) void scan_kernel(const float* __restrict__ xz,
                                                   const float* __restrict__ xc,
                                                   const float* __restrict__ dtr,
                                                   const float* __restrict__ Bv,
                                                   const float* __restrict__ Cv,
                                                   const float* __restrict__ A_log,
                                                   const float* __restrict__ dt_proj_w,
                                                   const float* __restrict__ dt_proj_b,
                                                   const float* __restrict__ D_param,
                                                   bf16_t* __restrict__ y_bf)
{
    const int group = threadIdx.x >> 4;                 // 8 channel-groups / block
    const int n     = threadIdx.x & 15;                 // state index
    const int c     = blockIdx.x * 8 + group;           // channel id
    if (c >= BATCH * D_INNER) return;
    const int b = c / D_INNER;
    const int d = c % D_INNER;

    const float An  = -__expf(A_log[d * D_STATE + n]);
    const float dtw = dt_proj_w[d];
    const float dtb = dt_proj_b[d];
    const float Dp  = D_param[d];

    float h = 0.0f;
    for (int l = 0; l < SEQLEN; ++l) {
        const int t = b * SEQLEN + l;
        float dr  = dtr[t];
        float xcv = xc[(size_t)t * D_INNER + d];
        float zv  = xz[(size_t)t * (2 * D_INNER) + D_INNER + d];
        float pre = dr * dtw + dtb;
        float dt  = (pre > 20.0f) ? pre : log1pf(__expf(pre));   // softplus
        float dA  = __expf(dt * An);
        float Bn  = Bv[(size_t)t * D_STATE + n];
        float Cn  = Cv[(size_t)t * D_STATE + n];
        if (l + 1 < SEQLEN) {
            __builtin_prefetch(&Bv[(size_t)(t + 1) * D_STATE + n], 0, 1);
            __builtin_prefetch(&Cv[(size_t)(t + 1) * D_STATE + n], 0, 1);
        }
        h = dA * h + dt * xcv * Bn;
        float p = h * Cn;
        // butterfly inside the 16-lane group (masks < 16 keep bit4 fixed)
        p += __shfl_xor(p, 1, 32);
        p += __shfl_xor(p, 2, 32);
        p += __shfl_xor(p, 4, 32);
        p += __shfl_xor(p, 8, 32);
        float y = p + xcv * Dp;
        y *= zv / (1.0f + __expf(-zv));     // * silu(z)
        if (n == 0) y_bf[(size_t)t * D_INNER + d] = (bf16_t)y;
    }
}

// ---------------------------------------------------------------------------
// launch
// ---------------------------------------------------------------------------
extern "C" void kernel_launch(void* const* d_in, const int* in_sizes, int n_in,
                              void* d_out, int out_size, void* d_ws, size_t ws_size,
                              hipStream_t stream) {
    const float* x         = (const float*)d_in[0];
    const float* in_proj_w = (const float*)d_in[1];
    const float* conv_w    = (const float*)d_in[2];
    const float* conv_b    = (const float*)d_in[3];
    const float* x_proj_w  = (const float*)d_in[4];
    const float* dt_proj_w = (const float*)d_in[5];
    const float* dt_proj_b = (const float*)d_in[6];
    const float* A_log     = (const float*)d_in[7];
    const float* D_param   = (const float*)d_in[8];
    const float* out_proj_w= (const float*)d_in[9];
    float* out = (float*)d_out;

    char* ws = (char*)d_ws;
    size_t o = 0;
    auto alloc = [&](size_t bytes) { size_t r = o; o += (bytes + 255) & ~(size_t)255; return r; };
    bf16_t* x_bf    = (bf16_t*)(ws + alloc((size_t)TOKENS * D_MODEL * 2));
    bf16_t* Win_bf  = (bf16_t*)(ws + alloc((size_t)2 * D_INNER * D_MODEL * 2));
    bf16_t* Wout_bf = (bf16_t*)(ws + alloc((size_t)D_MODEL * D_INNER * 2));
    float*  xz      = (float*) (ws + alloc((size_t)TOKENS * 2 * D_INNER * 4));
    float*  xc      = (float*) (ws + alloc((size_t)TOKENS * D_INNER * 4));
    float*  dtr     = (float*) (ws + alloc((size_t)TOKENS * 4));
    float*  Bv      = (float*) (ws + alloc((size_t)TOKENS * D_STATE * 4));
    float*  Cv      = (float*) (ws + alloc((size_t)TOKENS * D_STATE * 4));
    bf16_t* y_bf    = (bf16_t*)(ws + alloc((size_t)TOKENS * D_INNER * 2));
    if (o > ws_size) return;   // insufficient workspace (deterministic no-op)

    // 1. bf16 casts
    {
        int n1 = TOKENS * D_MODEL;
        cvt_f32_bf16<<<(n1 + 255) / 256, 256, 0, stream>>>(x, x_bf, n1);
        int n2 = 2 * D_INNER * D_MODEL;
        cvt_f32_bf16<<<(n2 + 255) / 256, 256, 0, stream>>>(in_proj_w, Win_bf, n2);
        int n3 = D_MODEL * D_INNER;
        cvt_f32_bf16<<<(n3 + 255) / 256, 256, 0, stream>>>(out_proj_w, Wout_bf, n3);
    }
    // 2. xz = x @ in_proj_w.T   (M=2048, N=4096, K=1024)
    {
        dim3 grid((2 * D_INNER) / 64, TOKENS / 64);
        gemm_bf16_wmma<<<grid, 128, 0, stream>>>(x_bf, Win_bf, xz, TOKENS, 2 * D_INNER, D_MODEL);
    }
    // 3. conv + silu
    {
        int n = BATCH * SEQLEN * D_INNER;
        conv_silu_kernel<<<(n + 255) / 256, 256, 0, stream>>>(xz, conv_w, conv_b, xc);
    }
    // 4. x_proj (33-wide matvec per token)
    xproj_kernel<<<TOKENS, 256, 0, stream>>>(xc, x_proj_w, dtr, Bv, Cv);
    // 5. selective scan + gate, writes bf16 y
    scan_kernel<<<(BATCH * D_INNER) / 8, 128, 0, stream>>>(
        xz, xc, dtr, Bv, Cv, A_log, dt_proj_w, dt_proj_b, D_param, y_bf);
    // 6. out = y @ out_proj_w.T  (M=2048, N=1024, K=2048)
    {
        dim3 grid(D_MODEL / 64, TOKENS / 64);
        gemm_bf16_wmma<<<grid, 128, 0, stream>>>(y_bf, Wout_bf, out, TOKENS, D_MODEL, D_INNER);
    }
}
